// MultiHeadedAttention_919123001945
// MI455X (gfx1250) — matrix operand order repaired and checked by reading, not compiled, not run on hardware
//
#include <hip/hip_runtime.h>
#include <hip/hip_bf16.h>

// ---------- types ----------
typedef __bf16 bf16;
typedef __attribute__((ext_vector_type(4)))  bf16     bf16x4;
typedef __attribute__((ext_vector_type(8)))  bf16     bf16x8;
typedef __attribute__((ext_vector_type(16))) bf16     v16bf;
typedef __attribute__((ext_vector_type(8)))  float    v8f;
typedef __attribute__((ext_vector_type(4)))  unsigned u32x4;
typedef __attribute__((ext_vector_type(8)))  int      i32x8;
typedef __attribute__((ext_vector_type(4)))  int      i32x4;

#define S_LEN  2048
#define DMODEL 1024
#define NHEAD  16
#define DHEAD  64

#if __has_builtin(__builtin_amdgcn_tensor_load_to_lds)
#define HAS_TDM 1
#else
#define HAS_TDM 0
#endif

static __device__ __forceinline__ v16bf cat8(bf16x8 lo, bf16x8 hi) {
    return __builtin_shufflevector(lo, hi, 0,1,2,3,4,5,6,7,8,9,10,11,12,13,14,15);
}

// ---------- TDM: 2D tile load, bf16 elements, optional LDS row padding ----------
// padc: pad_interval code (pad after 2^(padc+1) DWORDs), pada: pad_amount code ((pada+1) DWORDs)
static __device__ __forceinline__ void tdm_load_2d(
    unsigned lds_addr, const void* gptr,
    unsigned tensor_d0, unsigned tensor_d1, unsigned stride0,
    unsigned tile_d0, unsigned tile_d1, unsigned padc, unsigned pada)
{
#if HAS_TDM
    unsigned long long ga = (unsigned long long)(uintptr_t)gptr;
    u32x4 g0;
    g0[0] = 1u;                                     // count=1, user descriptor
    g0[1] = lds_addr;                               // LDS byte address (bits 63:32)
    g0[2] = (unsigned)ga;                           // global_addr[31:0]
    g0[3] = (unsigned)((ga >> 32) & 0x01FFFFFFu)    // global_addr[56:32]
          | (2u << 30);                             // type = 2 ("image")
    i32x8 g1;
    g1[0] = (int)((1u << 16)                        // data_size = 1 -> 2 bytes
                | (1u << 20)                        // pad_enable
                | (padc << 22) | (pada << 25));
    g1[1] = (int)(tensor_d0 << 16);                 // tensor_dim0[15:0] @ bits 63:48
    g1[2] = (int)((tensor_d0 >> 16) | (tensor_d1 << 16));
    g1[3] = (int)((tensor_d1 >> 16) | (tile_d0 << 16));
    g1[4] = (int)tile_d1;                           // tile_dim1 (tile_dim2 = 0)
    g1[5] = (int)stride0;                           // tensor_dim0_stride[31:0]
    g1[6] = 0;
    g1[7] = 0;
    i32x4 z4 = {0, 0, 0, 0};                        // groups 2/3: 2D tensor, unused dims = 0
    i32x8 z8 = {0, 0, 0, 0, 0, 0, 0, 0};
    __builtin_amdgcn_tensor_load_to_lds(g0, g1, z4, z4, z8, 0);
#else
    (void)lds_addr; (void)gptr; (void)tensor_d0; (void)tensor_d1;
    (void)stride0; (void)tile_d0; (void)tile_d1; (void)padc; (void)pada;
#endif
}

// ---------- f32 -> bf16 convert ----------
__global__ __launch_bounds__(256)
void cvt_f32_to_bf16(const float* __restrict__ in, bf16* __restrict__ out, int n4) {
    int i = blockIdx.x * 256 + threadIdx.x;
    if (i < n4) {
        float4 f = ((const float4*)in)[i];
        bf16x4 o;
        o[0] = (bf16)f.x; o[1] = (bf16)f.y; o[2] = (bf16)f.z; o[3] = (bf16)f.w;
        ((bf16x4*)out)[i] = o;
    }
}

// ---------- GEMM: Y[M,N] = A[M,K] @ W[N,K]^T + bias ----------
// MODE 0: bf16 out in [B,H,S,DK] head layout
// MODE 1: bf16 out in [B,H,DK,S] transposed-V layout
// MODE 2: f32 out plain [M,N]
#define LDP 40   // LDS pitch in halfs (80 B row = 64 B data + 16 B TDM pad; 16B aligned, conflict-free)

template<int MODE>
__global__ __launch_bounds__(256)
void gemm_bf16(const bf16* __restrict__ A, const bf16* __restrict__ W,
               const float* __restrict__ bias,
               bf16* __restrict__ outB, float* __restrict__ outF) {
    __shared__ bf16 As[2][128 * LDP];
    __shared__ bf16 Bs[2][128 * LDP];

    const int tid  = threadIdx.x;
    const int lane = tid & 31;
    const int wave = tid >> 5;
    const int wm   = wave & 3;
    const int wn   = wave >> 2;
    const int bm   = blockIdx.x * 128;
    const int bn   = blockIdx.y * 128;

    const int ln   = lane & 15;
    const int hiG  = lane >> 4;
    const int koff = hiG * 8;
    const int kgB  = hiG * 16;

    const bf16* Atile = A + (size_t)bm * DMODEL;
    const bf16* Wtile = W + (size_t)bn * DMODEL;

    auto stage = [&](int buf, int k0) {
#if HAS_TDM
        if (wave == 0) {
            // 128x32 bf16 tile, row 64B + 16B pad -> pitch 80B (padc=3: every 16 DW, pada=3: 4 DW)
            tdm_load_2d((unsigned)(uintptr_t)(void*)&As[buf][0], Atile + k0,
                        DMODEL, 128, DMODEL, 32, 128, 3, 3);
            tdm_load_2d((unsigned)(uintptr_t)(void*)&Bs[buf][0], Wtile + k0,
                        DMODEL, 128, DMODEL, 32, 128, 3, 3);
        }
#else
        #pragma unroll
        for (int i = 0; i < 2; ++i) {
            int c = tid + i * 256, row = c >> 2, col = (c & 3) * 8;
            *(bf16x8*)(&As[buf][row * LDP + col]) =
                *(const bf16x8*)(Atile + (size_t)row * DMODEL + k0 + col);
            *(bf16x8*)(&Bs[buf][row * LDP + col]) =
                *(const bf16x8*)(Wtile + (size_t)row * DMODEL + k0 + col);
        }
#endif
    };

    v8f acc[2][4] = {};
    stage(0, 0);

    for (int i = 0; i < 32; ++i) {
        const int cur = i & 1;
        if (i + 1 < 32) {
            stage(cur ^ 1, (i + 1) * 32);
#if HAS_TDM
            if (wave == 0) __builtin_amdgcn_s_wait_tensorcnt(2);  // current pair done (in-order)
#endif
        } else {
#if HAS_TDM
            if (wave == 0) __builtin_amdgcn_s_wait_tensorcnt(0);
#endif
        }
        __syncthreads();

        v16bf af[2], bfr[4];
        #pragma unroll
        for (int t = 0; t < 2; ++t) {
            const bf16* p = &As[cur][(wm * 32 + t * 16 + ln) * LDP];
            af[t] = cat8(*(const bf16x8*)(p + kgB), *(const bf16x8*)(p + kgB + 8));
        }
        #pragma unroll
        for (int t = 0; t < 4; ++t) {
            const bf16* p = &Bs[cur][(wn * 64 + t * 16 + ln) * LDP + kgB];
            bfr[t] = cat8(*(const bf16x8*)(p), *(const bf16x8*)(p + 8));
        }
        #pragma unroll
        for (int a = 0; a < 2; ++a)
            #pragma unroll
            for (int b = 0; b < 4; ++b)
                acc[a][b] = __builtin_amdgcn_wmma_f32_16x16x32_bf16(
                    false, af[a], false, bfr[b], (short)0, acc[a][b], false, false);
        __syncthreads();
    }

    #pragma unroll
    for (int i = 0; i < 2; ++i) {
        #pragma unroll
        for (int j = 0; j < 4; ++j) {
            #pragma unroll
            for (int r = 0; r < 8; ++r) {
                int gm = bm + wm * 32 + i * 16 + r + hiG * 8;
                int gn = bn + wn * 64 + j * 16 + ln;
                float v = acc[i][j][r] + bias[gn];
                if constexpr (MODE == 2) {
                    outF[(size_t)gm * DMODEL + gn] = v;
                } else {
                    int b = gm >> 11, s = gm & (S_LEN - 1);
                    int h = gn >> 6,  d = gn & (DHEAD - 1);
                    if constexpr (MODE == 0)
                        outB[(((size_t)(b * NHEAD + h) * S_LEN) + s) * DHEAD + d] = (bf16)v;
                    else
                        outB[(((size_t)(b * NHEAD + h) * DHEAD) + d) * S_LEN + s] = (bf16)v;
                }
            }
        }
    }
}

// ---------- causal flash attention ----------
// Q,K: [B*H, S, DK] bf16   V: [B*H, DK, S] bf16 (transposed)   ctx: [B*S, D] bf16
#define PLDP 72  // 144 B rows = 128 B data + 16 B TDM pad; 16B aligned, conflict-free

__global__ __launch_bounds__(128)
void attn_flash(const bf16* __restrict__ Qh, const bf16* __restrict__ Kh,
                const bf16* __restrict__ Vt, bf16* __restrict__ ctx) {
    const int qt   = blockIdx.x;
    const int bh   = blockIdx.y;
    const int lane = threadIdx.x & 31;
    const int wave = threadIdx.x >> 5;
    const int ln   = lane & 15;
    const int hiG  = lane >> 4;
    const int koff = hiG * 8;
    const int kg   = hiG * 16;
    const int q0   = qt * 64 + wave * 16;

    const bf16* Qbase = Qh + ((size_t)bh * S_LEN + q0) * DHEAD;
    const bf16* Kbase = Kh + (size_t)bh * S_LEN * DHEAD;
    const bf16* Vbase = Vt + (size_t)bh * DHEAD * S_LEN;

    __shared__ bf16 Ks[2][64 * PLDP];
    __shared__ bf16 Vs[2][64 * PLDP];
    __shared__ bf16 Ps[4][16 * PLDP];

    auto stageKV = [&](int buf, int kt) {
#if HAS_TDM
        if (wave == 0) {
            // K tile: 64 keys x 64 d (row 128B + 16B pad: padc=4, pada=3)
            tdm_load_2d((unsigned)(uintptr_t)(void*)&Ks[buf][0],
                        Kbase + (size_t)kt * 64 * DHEAD,
                        DHEAD, 64, DHEAD, 64, 64, 4, 3);
            // V tile: 64 d-rows x 64 keys, row stride S in the transposed tensor
            tdm_load_2d((unsigned)(uintptr_t)(void*)&Vs[buf][0],
                        Vbase + (size_t)kt * 64,
                        S_LEN, 64, S_LEN, 64, 64, 4, 3);
        }
#else
        #pragma unroll
        for (int i = 0; i < 4; ++i) {
            int c = (int)threadIdx.x + i * 128, row = c >> 3, col = (c & 7) * 8;
            *(bf16x8*)(&Ks[buf][row * PLDP + col]) =
                *(const bf16x8*)(Kbase + (size_t)(kt * 64 + row) * DHEAD + col);
            *(bf16x8*)(&Vs[buf][row * PLDP + col]) =
                *(const bf16x8*)(Vbase + (size_t)row * S_LEN + kt * 64 + col);
        }
#endif
    };

    // Q fragments stay in registers for the whole kernel
    v16bf qf[2];
    #pragma unroll
    for (int ks = 0; ks < 2; ++ks) {
        const bf16* p = Qbase + (size_t)ln * DHEAD + ks * 32;
        qf[ks] = cat8(*(const bf16x8*)(p + kg), *(const bf16x8*)(p + kg + 8));
    }

    float mrun[8], lrun[8];
    #pragma unroll
    for (int r = 0; r < 8; ++r) { mrun[r] = -1e30f; lrun[r] = 0.0f; }
    v8f oacc[4] = {};

    stageKV(0, 0);

    for (int kt = 0; kt <= qt; ++kt) {
        const int cur = kt & 1;
        if (kt < qt) {
            stageKV(cur ^ 1, kt + 1);
#if HAS_TDM
            if (wave == 0) __builtin_amdgcn_s_wait_tensorcnt(2);
#endif
        } else {
#if HAS_TDM
            if (wave == 0) __builtin_amdgcn_s_wait_tensorcnt(0);
#endif
        }
        __syncthreads();

        // ---- scores = Q @ K^T : batch 4 fragment loads, then 4 WMMAs per k-step ----
        v8f sacc[4] = {};
        #pragma unroll
        for (int ks = 0; ks < 2; ++ks) {
            v16bf kb[4];
            #pragma unroll
            for (int nt = 0; nt < 4; ++nt) {
                const bf16* p = &Ks[cur][(nt * 16 + ln) * PLDP + ks * 32 + kg];
                kb[nt] = cat8(*(const bf16x8*)(p), *(const bf16x8*)(p + 8));
            }
            #pragma unroll
            for (int nt = 0; nt < 4; ++nt)
                sacc[nt] = __builtin_amdgcn_wmma_f32_16x16x32_bf16(
                    false, qf[ks], false, kb[nt], (short)0, sacc[nt], false, false);
        }

        // ---- scale + causal mask ----
        float pv[4][8];
        #pragma unroll
        for (int nt = 0; nt < 4; ++nt)
            #pragma unroll
            for (int r = 0; r < 8; ++r) {
                float s   = sacc[nt][r] * 0.125f;     // 1/sqrt(64)
                int   row = q0 + r + hiG * 8;
                int   col = kt * 64 + nt * 16 + ln;
                pv[nt][r] = (col > row) ? -1e9f : s;
            }

        // ---- online softmax (butterfly over 16-lane groups, C-layout aware) ----
        #pragma unroll
        for (int r = 0; r < 8; ++r) {
            float t = fmaxf(fmaxf(pv[0][r], pv[1][r]), fmaxf(pv[2][r], pv[3][r]));
            #pragma unroll
            for (int off = 1; off < 16; off <<= 1)
                t = fmaxf(t, __shfl_xor(t, off, 32));
            float mnew  = fmaxf(mrun[r], t);
            float alpha = __expf(mrun[r] - mnew);
            mrun[r] = mnew;
            float rs = 0.0f;
            #pragma unroll
            for (int nt = 0; nt < 4; ++nt) {
                pv[nt][r] = __expf(pv[nt][r] - mnew);
                rs += pv[nt][r];
            }
            #pragma unroll
            for (int off = 1; off < 16; off <<= 1)
                rs += __shfl_xor(rs, off, 32);
            lrun[r] = lrun[r] * alpha + rs;
            #pragma unroll
            for (int dt = 0; dt < 4; ++dt)
                oacc[dt][r] *= alpha;
        }

        // ---- stage P (16x64 bf16) in this wave's LDS region ----
        #pragma unroll
        for (int nt = 0; nt < 4; ++nt)
            #pragma unroll
            for (int r = 0; r < 8; ++r)
                Ps[wave][(r + hiG * 8) * PLDP + nt * 16 + ln] = (bf16)pv[nt][r];
        __syncthreads();

        // ---- O += P @ V : batch 4 fragment loads, then 4 WMMAs per k-step ----
        #pragma unroll
        for (int ks = 0; ks < 2; ++ks) {
            const bf16* pp = &Ps[wave][ln * PLDP + ks * 32];
            v16bf pa = cat8(*(const bf16x8*)(pp + kg), *(const bf16x8*)(pp + kg + 8));
            v16bf vb[4];
            #pragma unroll
            for (int dt = 0; dt < 4; ++dt) {
                const bf16* vp = &Vs[cur][(dt * 16 + ln) * PLDP + ks * 32 + kg];
                vb[dt] = cat8(*(const bf16x8*)(vp), *(const bf16x8*)(vp + 8));
            }
            #pragma unroll
            for (int dt = 0; dt < 4; ++dt)
                oacc[dt] = __builtin_amdgcn_wmma_f32_16x16x32_bf16(
                    false, pa, false, vb[dt], (short)0, oacc[dt], false, false);
        }
        __syncthreads();
    }

    // ---- normalize + write context bf16 [B, S, H*DK] ----
    const int b = bh >> 4, h = bh & 15;
    #pragma unroll
    for (int r = 0; r < 8; ++r) {
        float inv = 1.0f / lrun[r];
        int   row = q0 + r + hiG * 8;
        #pragma unroll
        for (int dt = 0; dt < 4; ++dt)
            ctx[((size_t)b * S_LEN + row) * DMODEL + h * DHEAD + dt * 16 + ln] =
                (bf16)(oacc[dt][r] * inv);
    }
}

// ---------- launcher ----------
extern "C" void kernel_launch(void* const* d_in, const int* in_sizes, int n_in,
                              void* d_out, int out_size, void* d_ws, size_t ws_size,
                              hipStream_t stream) {
    const float* q  = (const float*)d_in[0];
    const float* k  = (const float*)d_in[1];
    const float* v  = (const float*)d_in[2];
    // d_in[3] = mask (unused: causal mask applied analytically)
    const float* Wq = (const float*)d_in[4];
    const float* bq = (const float*)d_in[5];
    const float* Wk = (const float*)d_in[6];
    const float* bk = (const float*)d_in[7];
    const float* Wv = (const float*)d_in[8];
    const float* bv = (const float*)d_in[9];
    const float* Wo = (const float*)d_in[10];
    const float* bo = (const float*)d_in[11];

    const size_t NX = (size_t)4 * S_LEN * DMODEL;   // 8388608
    const size_t NW = (size_t)DMODEL * DMODEL;      // 1048576

    bf16* Xq  = (bf16*)d_ws;
    bf16* Xk  = Xq  + NX;
    bf16* Xv  = Xk  + NX;
    bf16* Wqb = Xv  + NX;
    bf16* Wkb = Wqb + NW;
    bf16* Wvb = Wkb + NW;
    bf16* Wob = Wvb + NW;
    bf16* Qh  = Wob + NW;
    bf16* Kh  = Qh  + NX;
    bf16* Vt  = Kh  + NX;
    bf16* Ctx = Vt  + NX;

    int n4x = (int)(NX / 4), n4w = (int)(NW / 4);
    cvt_f32_to_bf16<<<(n4x + 255) / 256, 256, 0, stream>>>(q,  Xq,  n4x);
    cvt_f32_to_bf16<<<(n4x + 255) / 256, 256, 0, stream>>>(k,  Xk,  n4x);
    cvt_f32_to_bf16<<<(n4x + 255) / 256, 256, 0, stream>>>(v,  Xv,  n4x);
    cvt_f32_to_bf16<<<(n4w + 255) / 256, 256, 0, stream>>>(Wq, Wqb, n4w);
    cvt_f32_to_bf16<<<(n4w + 255) / 256, 256, 0, stream>>>(Wk, Wkb, n4w);
    cvt_f32_to_bf16<<<(n4w + 255) / 256, 256, 0, stream>>>(Wv, Wvb, n4w);
    cvt_f32_to_bf16<<<(n4w + 255) / 256, 256, 0, stream>>>(Wo, Wob, n4w);

    dim3 ggrid(64, 8);
    gemm_bf16<0><<<ggrid, 256, 0, stream>>>(Xq,  Wqb, bq, Qh,  nullptr);
    gemm_bf16<0><<<ggrid, 256, 0, stream>>>(Xk,  Wkb, bk, Kh,  nullptr);
    gemm_bf16<1><<<ggrid, 256, 0, stream>>>(Xv,  Wvb, bv, Vt,  nullptr);

    attn_flash<<<dim3(32, 64), 128, 0, stream>>>(Qh, Kh, Vt, Ctx);

    gemm_bf16<2><<<ggrid, 256, 0, stream>>>(Ctx, Wob, bo, nullptr, (float*)d_out);
}
